// DeformConv_31267361915085
// MI455X (gfx1250) — matrix operand order repaired and checked by reading, not compiled, not run on hardware
//
#include <hip/hip_runtime.h>
#include <hip/hip_bf16.h>
#include <stdint.h>

#define B_   4
#define CIN  256
#define COUT 256
#define H_   56
#define W_   56
#define HW   3136
#define KK   9

typedef __attribute__((ext_vector_type(16))) __bf16 v16bf;
typedef __attribute__((ext_vector_type(8)))  float  v8f;

__device__ __forceinline__ uint16_t f2bf(float f) {
    union { float f; uint32_t u; } v; v.f = f;
    uint32_t u = v.u;
    uint32_t r = u + 0x7fffu + ((u >> 16) & 1u);   // round-to-nearest-even
    return (uint16_t)(r >> 16);
}

__device__ __forceinline__ int iclamp(int v, int lo, int hi) {
    return v < lo ? lo : (v > hi ? hi : v);
}

// ---------------------------------------------------------------------------
// Kernel 0: pack weights  w[o][c][kk] (f32)  ->  Wt[kk][o][c] (bf16)
// ---------------------------------------------------------------------------
__global__ void wt_pack_kernel(const float* __restrict__ w, uint16_t* __restrict__ wt) {
    int t = blockIdx.x * blockDim.x + threadIdx.x;
    if (t >= COUT * CIN * KK) return;
    int c  = t & 255;
    int o  = (t >> 8) & 255;
    int kk = t >> 16;
    wt[t] = f2bf(w[(o * CIN + c) * KK + kk]);
}

// ---------------------------------------------------------------------------
// Kernel 1: offset conv  off[b][oc][ij] = conv3x3(x, w_off)[oc] + b_off[oc]
// blockIdx.y = b*18 + oc  (oc uniform per block -> w_off row staged in LDS)
// ---------------------------------------------------------------------------
__global__ __launch_bounds__(256)
void off_conv_kernel(const float* __restrict__ x,
                     const float* __restrict__ w_off,
                     const float* __restrict__ b_off,
                     float* __restrict__ off) {
    __shared__ float lw[CIN * KK];                 // 9216 B
    const int oc = blockIdx.y % (2 * KK);
    const int b  = blockIdx.y / (2 * KK);
    for (int i = threadIdx.x; i < CIN * KK; i += blockDim.x)
        lw[i] = w_off[oc * CIN * KK + i];
    __syncthreads();

    int ij = blockIdx.x * blockDim.x + threadIdx.x;
    if (ij >= HW) return;
    int yy = ij / W_, xx = ij % W_;
    const float* xb = x + (size_t)b * CIN * HW;

    float acc = b_off[oc];
    for (int c = 0; c < CIN; ++c) {
        const float* xc = xb + c * HW;
        const float* wr = lw + c * KK;
#pragma unroll
        for (int kh = 0; kh < 3; ++kh) {
            int y = yy + kh - 1;
            if ((unsigned)y >= (unsigned)H_) continue;
#pragma unroll
            for (int kw = 0; kw < 3; ++kw) {
                int xq = xx + kw - 1;
                if ((unsigned)xq >= (unsigned)W_) continue;
                acc = fmaf(xc[y * W_ + xq], wr[kh * 3 + kw], acc);
            }
        }
    }
    off[((size_t)b * (2 * KK) + oc) * HW + ij] = acc;
}

// ---------------------------------------------------------------------------
// Kernel 2: fused bilinear-sample + implicit GEMM via v_wmma_f32_16x16x32_bf16
// Tile: 64 pixels (M) x 128 out-channels (N); K = CIN*KK, chunked (kk, 32ch).
// 8 waves: 4 along M x 2 along N; wave = 16M x 64N = 4 accumulators.
// ---------------------------------------------------------------------------
__global__ __launch_bounds__(256)
void deform_wmma_kernel(const float* __restrict__ x,
                        const float* __restrict__ off,
                        const uint16_t* __restrict__ wt,
                        float* __restrict__ out) {
    __shared__ float4   s_w[64];          // per-pixel corner weights (masked)
    __shared__ int4     s_a[64];          // per-pixel clamped corner addrs
    __shared__ uint32_t s_A[64 * 16];     // A tile: 64 rows x 32 bf16 (swizzled)

    const int tid    = threadIdx.x;
    const int lane   = tid & 31;
    const int wave   = tid >> 5;
    const int wave_m = wave & 3;          // 0..3
    const int wave_n = wave >> 2;         // 0..1

    const int p_base  = blockIdx.x * 64;  // 3136 % 64 == 0 -> tile inside image
    const int b       = p_base / HW;
    const int ij_base = p_base % HW;
    const int n_base  = blockIdx.y * 128;

    const float* xb = x + (size_t)b * CIN * HW;

    v8f acc[4];
#pragma unroll
    for (int nt = 0; nt < 4; ++nt) {
        v8f z = {0.f, 0.f, 0.f, 0.f, 0.f, 0.f, 0.f, 0.f};
        acc[nt] = z;
    }

    const int m_s = tid & 63;             // sampling: pixel row
    const int cg  = tid >> 6;             // sampling: channel group (0..3)

    const int mrow = wave_m * 16 + (lane & 15);   // A-fragment row
    const int half = (lane >> 4) & 1;             // lane half selects K phase

    union Frag { uint32_t u[8]; v16bf v; };

    for (int kk = 0; kk < KK; ++kk) {
        __syncthreads();
        if (tid < 64) {
            int ij = ij_base + tid;
            int ii = ij / W_, jj = ij % W_;
            float dy = off[((size_t)b * (2 * KK) + 2 * kk) * HW + ij];
            float dx = off[((size_t)b * (2 * KK) + 2 * kk + 1) * HW + ij];
            int ky = kk / 3, kx = kk % 3;
            float py = dy + (float)(ky + ii - 1);
            float px = dx + (float)(kx + jj - 1);
            float y0f = floorf(py), x0f = floorf(px);
            float ty = py - y0f, tx = px - x0f;
            int y0 = (int)y0f, x0 = (int)x0f;
            int y1 = y0 + 1,  x1 = x0 + 1;
            float w00 = (1.f - ty) * (1.f - tx);
            float w01 = (1.f - ty) * tx;
            float w10 = ty * (1.f - tx);
            float w11 = ty * tx;
            bool vy0 = (unsigned)y0 < (unsigned)H_, vy1 = (unsigned)y1 < (unsigned)H_;
            bool vx0 = (unsigned)x0 < (unsigned)W_, vx1 = (unsigned)x1 < (unsigned)W_;
            int y0c = iclamp(y0, 0, H_ - 1), y1c = iclamp(y1, 0, H_ - 1);
            int x0c = iclamp(x0, 0, W_ - 1), x1c = iclamp(x1, 0, W_ - 1);
            s_w[tid] = make_float4((vy0 && vx0) ? w00 : 0.f,
                                   (vy0 && vx1) ? w01 : 0.f,
                                   (vy1 && vx0) ? w10 : 0.f,
                                   (vy1 && vx1) ? w11 : 0.f);
            s_a[tid] = make_int4(y0c * W_ + x0c, y0c * W_ + x1c,
                                 y1c * W_ + x0c, y1c * W_ + x1c);
        }
        __syncthreads();

        const uint16_t* wtk = wt + (size_t)kk * COUT * CIN;

        for (int c0 = 0; c0 < CIN; c0 += 32) {
            // ---- stage A tile: bilinear sample 8 channels per thread ----
            {
                float4 wv = s_w[m_s];
                int4   av = s_a[m_s];
                const float* xc = xb + (size_t)(c0 + cg * 8) * HW;
                uint32_t pk[4];
#pragma unroll
                for (int d = 0; d < 4; ++d) {
                    float v0 = wv.x * xc[av.x] + wv.y * xc[av.y]
                             + wv.z * xc[av.z] + wv.w * xc[av.w];
                    xc += HW;
                    float v1 = wv.x * xc[av.x] + wv.y * xc[av.y]
                             + wv.z * xc[av.z] + wv.w * xc[av.w];
                    xc += HW;
                    pk[d] = (uint32_t)f2bf(v0) | ((uint32_t)f2bf(v1) << 16);
                }
                int sw = cg ^ (m_s & 3);  // xor-swizzle to spread banks
                *((uint4*)&s_A[m_s * 16 + sw * 4]) =
                    make_uint4(pk[0], pk[1], pk[2], pk[3]);
            }
            __syncthreads();

            // ---- A fragment (ISA 16-bit A layout: lanes0-15 K0-7,16-23) ----
            Frag a;
            {
                int swl = (half)     ^ (mrow & 3);
                int swh = (half + 2) ^ (mrow & 3);
                uint4 lo = *((const uint4*)&s_A[mrow * 16 + swl * 4]);
                uint4 hi = *((const uint4*)&s_A[mrow * 16 + swh * 4]);
                a.u[0] = lo.x; a.u[1] = lo.y; a.u[2] = lo.z; a.u[3] = lo.w;
                a.u[4] = hi.x; a.u[5] = hi.y; a.u[6] = hi.z; a.u[7] = hi.w;
            }

            // ---- B fragments from global (L2-resident Wt) + WMMA ----
#pragma unroll
            for (int nt = 0; nt < 4; ++nt) {
                int n = n_base + wave_n * 64 + nt * 16 + (lane & 15);
                const uint16_t* wrow = wtk + (size_t)n * CIN + c0 + half * 8;
                Frag bf;
                uint4 q0 = *((const uint4*)wrow);
                uint4 q1 = *((const uint4*)(wrow + 16));
                bf.u[0] = q0.x; bf.u[1] = q0.y; bf.u[2] = q0.z; bf.u[3] = q0.w;
                bf.u[4] = q1.x; bf.u[5] = q1.y; bf.u[6] = q1.z; bf.u[7] = q1.w;
                acc[nt] = __builtin_amdgcn_wmma_f32_16x16x32_bf16(
                    false, a.v, false, bf.v, (short)0, acc[nt], false, false);
            }
            __syncthreads();
        }
    }

    // ---- epilogue: C layout -> contiguous pixels per lane, float4 stores ----
    int prow = ij_base + wave_m * 16 + ((lane >> 4) ? 8 : 0);
#pragma unroll
    for (int nt = 0; nt < 4; ++nt) {
        int o = n_base + wave_n * 64 + nt * 16 + (lane & 15);
        float* po = out + ((size_t)b * COUT + o) * HW + prow;
        float4 lo4 = make_float4(acc[nt][0], acc[nt][1], acc[nt][2], acc[nt][3]);
        float4 hi4 = make_float4(acc[nt][4], acc[nt][5], acc[nt][6], acc[nt][7]);
        *((float4*)po)       = lo4;
        *((float4*)(po + 4)) = hi4;
    }
}

// ---------------------------------------------------------------------------
extern "C" void kernel_launch(void* const* d_in, const int* in_sizes, int n_in,
                              void* d_out, int out_size, void* d_ws, size_t ws_size,
                              hipStream_t stream) {
    (void)in_sizes; (void)n_in; (void)out_size; (void)ws_size;
    const float* x     = (const float*)d_in[0];
    const float* w_off = (const float*)d_in[1];
    const float* b_off = (const float*)d_in[2];
    const float* w     = (const float*)d_in[3];
    float* out = (float*)d_out;

    // workspace: [ Wt bf16 : 1,179,648 B ][ off f32 : 903,168 B ]
    uint16_t* wt  = (uint16_t*)d_ws;
    float*    off = (float*)((char*)d_ws + (size_t)COUT * CIN * KK * sizeof(uint16_t));

    wt_pack_kernel<<<dim3((COUT * CIN * KK + 255) / 256), dim3(256), 0, stream>>>(w, wt);
    off_conv_kernel<<<dim3((HW + 255) / 256, B_ * 2 * KK), dim3(256), 0, stream>>>(
        x, w_off, b_off, off);
    deform_wmma_kernel<<<dim3(B_ * HW / 64, COUT / 128), dim3(256), 0, stream>>>(
        x, off, wt, out);
}
